// _FP8CheckpointWrapper_30494267801788
// MI455X (gfx1250) — matrix operand order repaired and checked by reading, not compiled, not run on hardware
//
#include <hip/hip_runtime.h>
#include <hip/hip_bf16.h>
#include <math.h>

// ---------------------------------------------------------------------------
// Fused MLP forward (x@W1+b1 -> exact GELU -> @W2+b2) + FP8-sim quantize of x.
// bf16 WMMA (v_wmma_f32_16x16x32_bf16), f32 accumulate, LDS double buffering
// with register-level global prefetch.
// ---------------------------------------------------------------------------

typedef __attribute__((ext_vector_type(16))) __bf16 v16bf;
typedef __attribute__((ext_vector_type(8)))  __bf16 v8bf;
typedef __attribute__((ext_vector_type(4)))  __bf16 v4bf;
typedef __attribute__((ext_vector_type(8)))  float  v8f;

#define BM 128
#define BN 128
#define BK 32
#define KPAD 8
#define LDT (BK + KPAD)   // LDS row stride in bf16 elements (80 B, 16B-aligned)

// A fragment (16x32 bf16, MxK): lane holds row m = lane&15.
// Elements 0..7 = K[k0..k0+7], 8..15 = K[k0+16..k0+23], k0 = lane<16 ? 0 : 8.
__device__ __forceinline__ v16bf load_a_frag(const __bf16* As, int row, int lane) {
  const int k0 = (lane < 16) ? 0 : 8;
  const __bf16* p = As + row * LDT + k0;
  v8bf lo = *(const v8bf*)(p);
  v8bf hi = *(const v8bf*)(p + 16);
  return __builtin_shufflevector(lo, hi, 0,1,2,3,4,5,6,7,8,9,10,11,12,13,14,15);
}

// B fragment (32x16 bf16, KxN), Bs transposed [col][k]: lane holds col n = lane&15.
// Elements 0..7 = K[kb..kb+7], 8..15 = K[kb+16..kb+23], kb = lane<16 ? 0 : 8.
__device__ __forceinline__ v16bf load_b_frag(const __bf16* Bs, int col, int lane) {
  const int kb = (lane < 16) ? 0 : 8;
  const __bf16* p = Bs + col * LDT + kb;
  v8bf lo = *(const v8bf*)(p);
  v8bf hi = *(const v8bf*)(p + 16);
  return __builtin_shufflevector(lo, hi, 0,1,2,3,4,5,6,7,8,9,10,11,12,13,14,15);
}

template<bool A_BF16, bool FUSE_GELU>
__global__ __launch_bounds__(256)
void mlp_gemm_wmma(const void* __restrict__ Av, const float* __restrict__ Bw,
                   const float* __restrict__ bias, void* __restrict__ Cv,
                   int M, int N, int K)
{
  __shared__ __align__(16) __bf16 As[2][BM * LDT];
  __shared__ __align__(16) __bf16 Bs[2][BN * LDT];

  const int t    = threadIdx.x;
  const int lane = t & 31;
  const int wave = t >> 5;
  const int mWave = (wave & 3) * 32;   // 4 waves along M
  const int nWave = (wave >> 2) * 64;  // 2 waves along N
  const int mBlk = blockIdx.y * BM;
  const int nBlk = blockIdx.x * BN;

  v8f acc[2][4];
  #pragma unroll
  for (int mi = 0; mi < 2; ++mi)
    #pragma unroll
    for (int ni = 0; ni < 4; ++ni) acc[mi][ni] = {};

  // staging assignments (256 threads)
  const int arow = t >> 1;          // A: 0..127, 16 cols each
  const int acol = (t & 1) * 16;    //    0 or 16
  const int bk4  = (t >> 5) * 4;    // B: 4x4 block, k base 0..28
  const int bn4  = (t & 31) * 4;    //    n base 0..124 (lanes -> 512B coalesced)

  // register staging for the in-flight tile
  float aR[16];      // f32 A path
  v8bf  aRH[2];      // bf16 A path
  float bR[16];      // B: 4 k-rows x 4 cols

  auto loadTile = [&](int kt) {
    if (A_BF16) {
      const __bf16* src = (const __bf16*)Av + (size_t)(mBlk + arow) * K + kt + acol;
      aRH[0] = *(const v8bf*)src;
      aRH[1] = *(const v8bf*)(src + 8);
    } else {
      const float* src = (const float*)Av + (size_t)(mBlk + arow) * K + kt + acol;
      #pragma unroll
      for (int i = 0; i < 16; i += 4)
        *(float4*)&aR[i] = *(const float4*)(src + i);
    }
    const float* bs = Bw + (size_t)(kt + bk4) * N + nBlk + bn4;
    #pragma unroll
    for (int j = 0; j < 4; ++j)
      *(float4*)&bR[j * 4] = *(const float4*)(bs + (size_t)j * N);
    if (kt + 2 * BK < K)   // global_prefetch_b8 for the tile after next (weights)
      __builtin_prefetch(bs + (size_t)BK * N, 0, 1);
  };

  auto storeTile = [&](int buf) {
    __bf16* a = &As[buf][arow * LDT + acol];
    if (A_BF16) {
      *(v8bf*)(a)     = aRH[0];
      *(v8bf*)(a + 8) = aRH[1];
    } else {
      #pragma unroll
      for (int i = 0; i < 16; ++i) a[i] = (__bf16)aR[i];   // packs to cvt_pk + b128
    }
    #pragma unroll
    for (int c = 0; c < 4; ++c) {   // transpose 4x4 block: one b64 store per column
      v4bf w = { (__bf16)bR[0 * 4 + c], (__bf16)bR[1 * 4 + c],
                 (__bf16)bR[2 * 4 + c], (__bf16)bR[3 * 4 + c] };
      *(v4bf*)&Bs[buf][(bn4 + c) * LDT + bk4] = w;
    }
  };

  const int nTiles = K / BK;

  loadTile(0);
  storeTile(0);
  __syncthreads();

  for (int ti = 0; ti < nTiles; ++ti) {
    const int buf = ti & 1;
    const bool more = (ti + 1) < nTiles;
    if (more) loadTile((ti + 1) * BK);   // issue next tile's global loads early

    // ---- compute: each wave does a 32x64 patch = 2x4 wmma per k-step ----
    v16bf afr[2];
    #pragma unroll
    for (int mi = 0; mi < 2; ++mi)
      afr[mi] = load_a_frag(&As[buf][0], mWave + mi * 16 + (lane & 15), lane);
    #pragma unroll
    for (int ni = 0; ni < 4; ++ni) {
      v16bf bfr = load_b_frag(&Bs[buf][0], nWave + ni * 16 + (lane & 15), lane);
      #pragma unroll
      for (int mi = 0; mi < 2; ++mi)
        acc[mi][ni] = __builtin_amdgcn_wmma_f32_16x16x32_bf16(
            false, afr[mi], false, bfr, (short)0, acc[mi][ni], false, false);
    }

    if (more) storeTile(buf ^ 1);        // convert + stage into the other buffer
    __syncthreads();                     // one barrier per k-step
  }

  // ---- epilogue: C layout m = r + (lane<16?0:8), n = lane&15 per 16x16 tile ----
  #pragma unroll
  for (int mi = 0; mi < 2; ++mi) {
    const int mBase = mBlk + mWave + mi * 16 + ((lane < 16) ? 0 : 8);
    #pragma unroll
    for (int ni = 0; ni < 4; ++ni) {
      const int nG = nBlk + nWave + ni * 16 + (lane & 15);
      const float bv = bias[nG];
      #pragma unroll
      for (int r = 0; r < 8; ++r) {
        float v = acc[mi][ni][r] + bv;
        if (FUSE_GELU) {
          v = 0.5f * v * (1.0f + erff(v * 0.70710678118654752f));  // exact GELU
          ((__bf16*)Cv)[(size_t)(mBase + r) * N + nG] = (__bf16)v;
        } else {
          ((float*)Cv)[(size_t)(mBase + r) * N + nG] = v;
        }
      }
    }
  }
}

// ---------------------------------------------------------------------------
// amax over |x| via block reduce + atomicMax on non-negative float bits
// ---------------------------------------------------------------------------
__global__ __launch_bounds__(256)
void amax_kernel(const float* __restrict__ x, unsigned* __restrict__ amax_bits, size_t n) {
  __shared__ float red[256];
  float m = 0.0f;
  for (size_t i = (size_t)blockIdx.x * 256u + threadIdx.x; i < n;
       i += (size_t)gridDim.x * 256u)
    m = fmaxf(m, fabsf(x[i]));
  red[threadIdx.x] = m;
  __syncthreads();
  for (int s = 128; s > 0; s >>= 1) {
    if ((int)threadIdx.x < s)
      red[threadIdx.x] = fmaxf(red[threadIdx.x], red[threadIdx.x + s]);
    __syncthreads();
  }
  if (threadIdx.x == 0) atomicMax(amax_bits, __float_as_uint(red[0]));
}

// ---------------------------------------------------------------------------
// q = (float)(int16)clamp(x*scale, +-448); scale = 448/max(amax, 1e-12)
// ---------------------------------------------------------------------------
__global__ __launch_bounds__(256)
void quantize_kernel(const float* __restrict__ x, const unsigned* __restrict__ amax_bits,
                     float* __restrict__ q, float* __restrict__ scale_out, size_t n) {
  const float amax  = fmaxf(__uint_as_float(*amax_bits), 1e-12f);
  const float scale = 448.0f / amax;
  const size_t i0 = (size_t)blockIdx.x * 256u + threadIdx.x;
  if (i0 == 0) *scale_out = scale;
  for (size_t i = i0; i < n; i += (size_t)gridDim.x * 256u) {
    float v = fminf(fmaxf(x[i] * scale, -448.0f), 448.0f);
    q[i] = (float)(short)v;   // int16 cast truncates toward zero, like jnp astype
  }
}

// ---------------------------------------------------------------------------
extern "C" void kernel_launch(void* const* d_in, const int* in_sizes, int n_in,
                              void* d_out, int out_size, void* d_ws, size_t ws_size,
                              hipStream_t stream) {
  (void)in_sizes; (void)n_in; (void)out_size; (void)ws_size;

  const float* x  = (const float*)d_in[0];   // [4,4096,2048] f32
  const float* W1 = (const float*)d_in[1];   // [2048,8192]
  const float* b1 = (const float*)d_in[2];   // [8192]
  const float* W2 = (const float*)d_in[3];   // [8192,2048]
  const float* b2 = (const float*)d_in[4];   // [2048]

  const int D = 2048, F = 8192;
  const int M = 4 * 4096;                    // 16384 rows
  const size_t nx = (size_t)M * D;           // 33,554,432

  float* out       = (float*)d_out;          // [M,D] f32
  float* qout      = out + nx;               // [M,D] int16 values stored as f32
  float* scale_out = qout + nx;              // scalar

  unsigned* amax_bits = (unsigned*)d_ws;               // 4 bytes
  void* h = (void*)((char*)d_ws + 256);                // bf16 h [M,F] = 256 MiB

  hipMemsetAsync(d_ws, 0, 4, stream);                  // zero amax accumulator
  amax_kernel<<<2048, 256, 0, stream>>>(x, amax_bits, nx);

  dim3 g1(F / BN, M / BM);  // (64, 128)
  mlp_gemm_wmma<false, true><<<g1, 256, 0, stream>>>(x, W1, b1, h, M, F, D);

  dim3 g2(D / BN, M / BM);  // (16, 128)
  mlp_gemm_wmma<true, false><<<g2, 256, 0, stream>>>(h, W2, b2, out, M, D, F);

  quantize_kernel<<<2048, 256, 0, stream>>>(x, amax_bits, qout, scale_out, nx);
}
